// LigandGCN_48249662603679
// MI455X (gfx1250) — matrix operand order repaired and checked by reading, not compiled, not run on hardware
//
#include <hip/hip_runtime.h>
#include <hip/hip_bf16.h>

#define N_NODES  100000
#define N_EDGES  1600000
#define N_GRAPHS 4096
#define IN_DIM   14
#define HID      128

typedef __attribute__((ext_vector_type(16))) _Float16 v16h;
typedef __attribute__((ext_vector_type(8)))  _Float16 v8h;
typedef __attribute__((ext_vector_type(8)))  float    v8f;

// ---------------------------------------------------------------- utilities
__global__ void zero_f32_kernel(float* __restrict__ p, size_t n) {
    size_t i = (size_t)blockIdx.x * blockDim.x + threadIdx.x;
    size_t stride = (size_t)gridDim.x * blockDim.x;
    for (; i < n; i += stride) p[i] = 0.0f;
}

// ------------------------------------------------ layer-1 scatter (14-wide)
__global__ void scatter14_kernel(const float* __restrict__ x,
                                 const int* __restrict__ src,
                                 const int* __restrict__ dst,
                                 float* __restrict__ agg) {
    int e = blockIdx.x * blockDim.x + threadIdx.x;
    if (e >= N_EDGES) return;
    int s = src[e], d = dst[e];
    const float* xs = x + (size_t)s * IN_DIM;
    float* ag = agg + (size_t)d * IN_DIM;
    #pragma unroll
    for (int k = 0; k < IN_DIM; ++k) atomicAdd(&ag[k], xs[k]);
}

// ------------- fused: t1 = relu((x + agg14) @ W1a + b1a), stored as f16
__global__ __launch_bounds__(128) void lin1_kernel(
        const float* __restrict__ x, const float* __restrict__ agg,
        const float* __restrict__ W, const float* __restrict__ b,
        _Float16* __restrict__ t1) {
    __shared__ float Ws[IN_DIM * HID];
    __shared__ float bs[HID];
    int tid = threadIdx.x;
    for (int i = tid; i < IN_DIM * HID; i += 128) Ws[i] = W[i];
    bs[tid] = b[tid];
    __syncthreads();
    for (int n = blockIdx.x; n < N_NODES; n += gridDim.x) {
        float h0[IN_DIM];
        #pragma unroll
        for (int k = 0; k < IN_DIM; ++k)
            h0[k] = x[(size_t)n * IN_DIM + k] + agg[(size_t)n * IN_DIM + k];
        float acc = bs[tid];
        #pragma unroll
        for (int k = 0; k < IN_DIM; ++k)
            acc = fmaf(h0[k], Ws[k * HID + tid], acc);
        t1[(size_t)n * HID + tid] = (_Float16)fmaxf(acc, 0.0f);
    }
}

// ---- pre-swizzle 128x128 f32 weights into per-lane WMMA B-fragment order
// B operand (16-bit, 32x16 tile): lanes 0-15 hold K=0..7 / K=16..23 of column N=lane,
// lanes 16-31 hold K=8..15 / K=24..31 of column N=lane-16; slot j within lane <-> K.
// Fragment index: [ks (K/32 step)][ct (N/16 tile)][lane][j] contiguous.
__global__ void prep_w_kernel(const float* __restrict__ W1b,
                              const float* __restrict__ W2a,
                              const float* __restrict__ W2b,
                              const float* __restrict__ Wout,
                              _Float16* __restrict__ wf) {
    int idx = blockIdx.x * blockDim.x + threadIdx.x;
    if (idx >= 4 * 16384) return;
    int m = idx >> 14, o = idx & 16383;
    const float* W = (m == 0) ? W1b : (m == 1) ? W2a : (m == 2) ? W2b : Wout;
    int j    =  o        & 15;
    int lane = (o >> 4)  & 31;
    int ct   = (o >> 9)  & 7;
    int ks   = (o >> 12) & 3;
    int k = ks * 32 + (j & 8) * 2 + (lane >> 4) * 8 + (j & 7);
    int c = ct * 16 + (lane & 15);
    wf[idx] = (_Float16)W[k * HID + c];
}

// ------------------- WMMA GEMM: D[rows x 128] = act(A[rows x 128] @ W + b)
// B staged once per block in LDS (32 KB of 320 KB/WGP), shared by 8 waves.
// One wave = two 16-row tiles (32 rows) x 128 cols: each B fragment feeds
// two back-to-back v_wmma; 8 col-tiles x 4 K-steps x 2 tiles = 64 wmma/wave.
// REQUIRES rows % 32 == 0 (holds: 100000 and 4096), so the epilogue is
// branch-free straight-line code (EXEC stays all-1s throughout).
template <int RELU, int WRITE_F32, int WRITE_F16>
__global__ __launch_bounds__(256) void gemm128_wmma_kernel(
        const _Float16* __restrict__ A, const _Float16* __restrict__ Bf,
        const float* __restrict__ bias, int rows,
        float* __restrict__ outF, _Float16* __restrict__ outH) {
    __shared__ _Float16 Bs[4 * 8 * 32 * 16];   // 16384 f16 = 32 KB

    // cooperative copy: 32 KB / 256 threads = 128 B (8 x uint4) per thread
    {
        const uint4* gsrc = (const uint4*)Bf;
        uint4* ldst = (uint4*)Bs;
        #pragma unroll
        for (int i = 0; i < 8; ++i)
            ldst[threadIdx.x + i * 256] = gsrc[threadIdx.x + i * 256];
    }
    __syncthreads();

    const int lane  = threadIdx.x & 31;
    const int wave  = threadIdx.x >> 5;
    const int rowBase = (blockIdx.x * 8 + wave) * 32;   // 32 rows per wave
    if (rowBase >= rows) return;             // wave-uniform: EXEC stays all-1s
    const int mrow  = lane & 15;
    const int khalf = lane >> 4;

    // hoist all A fragments: 2 row-tiles x 4 K-steps (64 VGPRs)
    // A 16-bit 16x32 layout: lanes 0-15: K=0..7 / K=16..23,
    //                        lanes 16-31: K=8..15 / K=24..31
    v16h afr[2][4];
    #pragma unroll
    for (int m = 0; m < 2; ++m) {
        const _Float16* Arow = A + (size_t)(rowBase + m * 16 + mrow) * HID;
        #pragma unroll
        for (int ks = 0; ks < 4; ++ks) {
            v8h a0 = *(const v8h*)(Arow + ks * 32 + khalf * 8);
            v8h a1 = *(const v8h*)(Arow + ks * 32 + 16 + khalf * 8);
            afr[m][ks] = __builtin_shufflevector(a0, a1,
                                                 0, 1, 2, 3, 4, 5, 6, 7,
                                                 8, 9, 10, 11, 12, 13, 14, 15);
        }
    }

    // per-lane output row origin: lanes 0-15 -> M=r, lanes 16-31 -> M=8+r
    const size_t orow0 = (size_t)(rowBase + khalf * 8) * HID;

    #pragma unroll
    for (int ct = 0; ct < 8; ++ct) {
        v8f acc0 = {}, acc1 = {};
        #pragma unroll
        for (int ks = 0; ks < 4; ++ks) {
            v16h bfr = *(const v16h*)(Bs + ((ks * 8 + ct) * 32 + lane) * 16);
            acc0 = __builtin_amdgcn_wmma_f32_16x16x32_f16(
                       false, afr[0][ks], false, bfr, (short)0, acc0, false, false);
            acc1 = __builtin_amdgcn_wmma_f32_16x16x32_f16(
                       false, afr[1][ks], false, bfr, (short)0, acc1, false, false);
        }
        const int col = ct * 16 + mrow;
        const float bv = bias[col];
        float*    pF = WRITE_F32 ? (outF + orow0 + col) : nullptr;
        _Float16* pH = WRITE_F16 ? (outH + orow0 + col) : nullptr;
        #pragma unroll
        for (int r = 0; r < 8; ++r) {
            float v0 = acc0[r] + bv;
            float v1 = acc1[r] + bv;
            if (RELU) { v0 = fmaxf(v0, 0.0f); v1 = fmaxf(v1, 0.0f); }
            if (WRITE_F32) {
                pF[(size_t)r * HID] = v0;
                pF[(size_t)(r + 16) * HID] = v1;
            }
            if (WRITE_F16) {
                pH[(size_t)r * HID] = (_Float16)v0;
                pH[(size_t)(r + 16) * HID] = (_Float16)v1;
            }
        }
    }
}

// --------------------------- layer-2 scatter (128-wide, float4 per thread)
__global__ void scatter128_kernel(const float* __restrict__ h,
                                  const int* __restrict__ src,
                                  const int* __restrict__ dst,
                                  float* __restrict__ agg) {
    long t = (long)blockIdx.x * blockDim.x + threadIdx.x;
    if (t >= (long)N_EDGES * 32) return;
    int e = (int)(t >> 5);
    int q = ((int)t & 31) << 2;
    int s = src[e], d = dst[e];
    const float4 v = *(const float4*)(h + (size_t)s * HID + q);
    float* ag = agg + (size_t)d * HID + q;
    atomicAdd(ag + 0, v.x);
    atomicAdd(ag + 1, v.y);
    atomicAdd(ag + 2, v.z);
    atomicAdd(ag + 3, v.w);
}

// h1s = h1 + agg2, downconvert to f16 for the next WMMA A operand
__global__ void addcvt_kernel(const float* __restrict__ a,
                              const float* __restrict__ b,
                              _Float16* __restrict__ o, size_t n) {
    size_t i = (size_t)blockIdx.x * blockDim.x + threadIdx.x;
    if (i < n) o[i] = (_Float16)(a[i] + b[i]);
}

// --------------------------------------------------- mean-pool via atomics
__global__ void pool_kernel(const float* __restrict__ h,
                            const int* __restrict__ batch,
                            float* __restrict__ sums, float* __restrict__ cnts) {
    int t = blockIdx.x * blockDim.x + threadIdx.x;
    if (t >= N_NODES * HID) return;
    int n = t >> 7, c = t & 127;
    int g = batch[n];
    atomicAdd(&sums[(size_t)g * HID + c], h[t]);
    if (c == 0) atomicAdd(&cnts[g], 1.0f);
}

__global__ void mean_kernel(const float* __restrict__ sums,
                            const float* __restrict__ cnts,
                            _Float16* __restrict__ mh) {
    int t = blockIdx.x * blockDim.x + threadIdx.x;
    if (t >= N_GRAPHS * HID) return;
    int g = t >> 7;
    float c = cnts[g];
    c = (c > 1.0f) ? c : 1.0f;
    mh[t] = (_Float16)(sums[t] / c);
}

// --------------------------------------------------------------- launcher
extern "C" void kernel_launch(void* const* d_in, const int* in_sizes, int n_in,
                              void* d_out, int out_size, void* d_ws, size_t ws_size,
                              hipStream_t stream) {
    (void)in_sizes; (void)n_in; (void)out_size; (void)ws_size;
    const float* x    = (const float*)d_in[0];
    const int*   ei   = (const int*)d_in[1];
    const int*   batch= (const int*)d_in[2];
    const float* W1a  = (const float*)d_in[3];
    const float* b1a  = (const float*)d_in[4];
    const float* W1b  = (const float*)d_in[5];
    const float* b1b  = (const float*)d_in[6];
    const float* W2a  = (const float*)d_in[7];
    const float* b2a  = (const float*)d_in[8];
    const float* W2b  = (const float*)d_in[9];
    const float* b2b  = (const float*)d_in[10];
    const float* Wout = (const float*)d_in[11];
    const float* bout = (const float*)d_in[12];
    const int* src = ei;
    const int* dst = ei + N_EDGES;

    // carve workspace (~131 MB; whole working set is L2-resident on MI455X)
    char* ws = (char*)d_ws;
    size_t off = 0;
    auto carve = [&](size_t bytes) -> char* {
        char* p = ws + off;
        off += (bytes + 255) & ~(size_t)255;
        return p;
    };
    float*    bufP = (float*)   carve((size_t)N_NODES * HID * 4);  // f32 node buf
    float*    bufQ = (float*)   carve((size_t)N_NODES * HID * 4);  // agg buf
    _Float16* bufH = (_Float16*)carve((size_t)N_NODES * HID * 2);  // f16 activations
    _Float16* wf   = (_Float16*)carve((size_t)4 * 16384 * 2);      // swizzled weights
    float*    sums = (float*)   carve((size_t)N_GRAPHS * HID * 4);
    float*    cnts = (float*)   carve((size_t)N_GRAPHS * 4);
    _Float16* mh   = (_Float16*)carve((size_t)N_GRAPHS * HID * 2);

    const int gemmBlocksN = (N_NODES + 255) / 256;   // 256 rows per block

    // 0) swizzle W1b/W2a/W2b/Wout into WMMA B-fragment order (f16)
    prep_w_kernel<<<(4 * 16384 + 255) / 256, 256, 0, stream>>>(W1b, W2a, W2b, Wout, wf);

    // 1) agg14 = scatter_add(x[src] -> dst)
    zero_f32_kernel<<<2048, 256, 0, stream>>>(bufQ, (size_t)N_NODES * IN_DIM);
    scatter14_kernel<<<(N_EDGES + 255) / 256, 256, 0, stream>>>(x, src, dst, bufQ);

    // 2) t1 = relu((x+agg14) @ W1a + b1a)  -> f16
    lin1_kernel<<<4096, 128, 0, stream>>>(x, bufQ, W1a, b1a, bufH);

    // 3) h1 = relu(t1 @ W1b + b1b)  -> f32
    gemm128_wmma_kernel<1, 1, 0><<<gemmBlocksN, 256, 0, stream>>>(
        bufH, wf + 0 * 16384, b1b, N_NODES, bufP, (_Float16*)nullptr);

    // 4) agg2 = scatter_add(h1[src] -> dst)
    zero_f32_kernel<<<4096, 256, 0, stream>>>(bufQ, (size_t)N_NODES * HID);
    scatter128_kernel<<<(int)(((long)N_EDGES * 32 + 255) / 256), 256, 0, stream>>>(
        bufP, src, dst, bufQ);

    // 5) h1s = h1 + agg2 -> f16
    addcvt_kernel<<<(N_NODES * HID + 255) / 256, 256, 0, stream>>>(
        bufP, bufQ, bufH, (size_t)N_NODES * HID);

    // 6) t2 = relu(h1s @ W2a + b2a) -> f16 (in-place per-wave, rows%32==0)
    gemm128_wmma_kernel<1, 0, 1><<<gemmBlocksN, 256, 0, stream>>>(
        bufH, wf + 1 * 16384, b2a, N_NODES, (float*)nullptr, bufH);

    // 7) h2 = relu(t2 @ W2b + b2b) -> f32
    gemm128_wmma_kernel<1, 1, 0><<<gemmBlocksN, 256, 0, stream>>>(
        bufH, wf + 2 * 16384, b2b, N_NODES, bufP, (_Float16*)nullptr);

    // 8) pooled mean per graph
    zero_f32_kernel<<<512, 256, 0, stream>>>(sums, (size_t)N_GRAPHS * HID);
    zero_f32_kernel<<<16, 256, 0, stream>>>(cnts, (size_t)N_GRAPHS);
    pool_kernel<<<(N_NODES * HID + 255) / 256, 256, 0, stream>>>(bufP, batch, sums, cnts);
    mean_kernel<<<(N_GRAPHS * HID + 255) / 256, 256, 0, stream>>>(sums, cnts, mh);

    // 9) out = pooled @ Wout + bout -> f32 d_out
    gemm128_wmma_kernel<0, 1, 0><<<(N_GRAPHS + 255) / 256, 256, 0, stream>>>(
        mh, wf + 3 * 16384, bout, N_GRAPHS, (float*)d_out, (_Float16*)nullptr);
}
